// GPT_48447231099450
// MI455X (gfx1250) — compile-verified, matrix operand order checked by reading, not run on hardware
//
#include <hip/hip_runtime.h>
#include <hip/hip_bf16.h>
#include <math.h>
#include <stdint.h>

// GPT forward for MI455X (gfx1250): f16 WMMA GEMMs with f32 accumulate,
// async direct-to-LDS double-buffered tiles, flash-attention with online
// softmax on WMMA fragments.

typedef _Float16 half8 __attribute__((ext_vector_type(8)));
typedef _Float16 v16h  __attribute__((ext_vector_type(16)));
typedef float    v8f   __attribute__((ext_vector_type(8)));

#define BB   8
#define TT   1024
#define EE   1024
#define HH   16
#define DD   64
#define FF_  4096
#define MTOT 8192   // B*T

// ---------------------------------------------------------------- fragments

// A-fragment (16x32, f16) per CDNA5 ISA layout; also used for B as the
// A-layout of B^T (lane r = output column, contiguous K in memory).
__device__ __forceinline__ v16h load_frag(const _Float16* __restrict__ base, int ld) {
  const int lane = threadIdx.x & 31;
  const int r    = lane & 15;
  const int kg   = lane >> 4;               // 0: K base 0, 1: K base 8
  const _Float16* p = base + r * ld + kg * 8;
  half8 lo = *(const half8*)(p);            // K = kg*8 .. +7
  half8 hi = *(const half8*)(p + 16);       // K = 16 + kg*8 .. +7
  v16h out;
#pragma unroll
  for (int i = 0; i < 8; ++i) { out[i] = lo[i]; out[i + 8] = hi[i]; }
  return out;
}

__device__ __forceinline__ v8f wmma32(v16h a, v16h b, v8f c) {
  return __builtin_amdgcn_wmma_f32_16x16x32_f16(false, a, false, b, (short)0, c,
                                                false, false);
}

// Async global->LDS copy of 32 bytes (two b128) per lane; ASYNCcnt tracked.
__device__ __forceinline__ void async_copy32(uint32_t lds_off, const _Float16* g) {
  uint64_t ga = (uint64_t)(uintptr_t)g;
  asm volatile("global_load_async_to_lds_b128 %0, %1, off\n\t"
               "global_load_async_to_lds_b128 %0, %1, off offset:16"
               :: "v"(lds_off), "v"(ga) : "memory");
}

__device__ __forceinline__ void wait_async0() {
  asm volatile("s_wait_asynccnt 0x0" ::: "memory");
}

// ---------------------------------------------------------------- pre/post

__global__ void patch_pca_kernel(const float* __restrict__ img,
                                 const float* __restrict__ mmean,
                                 const float* __restrict__ mcomp,
                                 float* __restrict__ coef) {
  int id = blockIdx.x * 256 + threadIdx.x;        // over 8192*32
  int t = id >> 5, c = id & 31;
  int b = t >> 10, idx = t & 1023;
  int hh = idx >> 5, ww = idx & 31;
  const float* base = img + ((size_t)b * 256) * 256;
  float acc = 0.f;
#pragma unroll
  for (int p1 = 0; p1 < 8; ++p1)
#pragma unroll
    for (int p2 = 0; p2 < 8; ++p2) {
      int j = p1 * 8 + p2;
      float pix = base[(size_t)(hh * 8 + p1) * 256 + (ww * 8 + p2)];
      acc += (pix - mmean[j]) * mcomp[c * 64 + j];
    }
  coef[(size_t)t * 32 + c] = acc;
}

__global__ void embed_kernel(const float* __restrict__ coef,
                             const float* __restrict__ tokw,
                             const float* __restrict__ pos,
                             float* __restrict__ x) {
  int id = blockIdx.x * 256 + threadIdx.x;        // over 8192*1024
  int t = id >> 10, e = id & 1023;
  float acc = 0.f;
#pragma unroll
  for (int c = 0; c < 32; ++c)
    acc += coef[(size_t)t * 32 + c] * tokw[(size_t)e * 32 + c];
  x[id] = acc + pos[(size_t)(t & 1023) * 1024 + e];
}

__global__ void f32tof16_kernel(const float* __restrict__ in,
                                _Float16* __restrict__ out, int n) {
  int i = blockIdx.x * 256 + threadIdx.x;
  if (i < n) out[i] = (_Float16)in[i];
}

__global__ __launch_bounds__(256)
void ln_kernel(const float* __restrict__ x, const float* __restrict__ g,
               const float* __restrict__ b, _Float16* __restrict__ out) {
  const int row = blockIdx.x;
  const float* xr = x + (size_t)row * EE;
  float s = 0.f, ss = 0.f;
  float v[4];
#pragma unroll
  for (int i = 0; i < 4; ++i) {
    v[i] = xr[threadIdx.x + 256 * i];
    s += v[i]; ss += v[i] * v[i];
  }
#pragma unroll
  for (int m = 16; m >= 1; m >>= 1) { s += __shfl_xor(s, m, 32); ss += __shfl_xor(ss, m, 32); }
  __shared__ float rs_[8], rss_[8];
  int w = threadIdx.x >> 5;
  if ((threadIdx.x & 31) == 0) { rs_[w] = s; rss_[w] = ss; }
  __syncthreads();
  s = 0.f; ss = 0.f;
#pragma unroll
  for (int i = 0; i < 8; ++i) { s += rs_[i]; ss += rss_[i]; }
  float mu = s * (1.f / EE);
  float var = ss * (1.f / EE) - mu * mu;
  float rstd = rsqrtf(var + 1e-5f);
#pragma unroll
  for (int i = 0; i < 4; ++i) {
    int c = threadIdx.x + 256 * i;
    out[(size_t)row * EE + c] = (_Float16)((v[i] - mu) * rstd * g[c] + b[c]);
  }
}

// ---------------------------------------------------------------- WMMA GEMM
// C[M,N] = epilogue(A[M,K] * W[N,K]^T + bias[N]); M = 8192 fixed.
// MODE 0: outH = f16(acc+bias); MODE 1: outF = resid + acc + bias;
// MODE 2: outH = f16(gelu(acc+bias)).
// Tiles stream global->LDS via GLOBAL_LOAD_ASYNC_TO_LDS_B128 with double
// buffering: the async copy of tile k+1 overlaps the 8 WMMAs of tile k.

#define BM  128
#define BN  128
#define BK  32
#define LDA 40   // padded halves (80B rows, 16B aligned)

template <int MODE>
__global__ __launch_bounds__(256)
void gemm_kernel(const _Float16* __restrict__ A, const _Float16* __restrict__ W,
                 const float* __restrict__ bias, const float* __restrict__ resid,
                 float* __restrict__ outF, _Float16* __restrict__ outH,
                 int K, int N) {
  __shared__ __align__(16) _Float16 As[2][BM * LDA];
  __shared__ __align__(16) _Float16 Bs[2][BN * LDA];
  const int tid = threadIdx.x;
  const int m0 = blockIdx.x * BM;
  const int n0 = blockIdx.y * BN;
  const int w  = tid >> 5;
  const int wm = (w & 3) * 32;     // 4 waves along M
  const int wn = (w >> 2) * 64;    // 2 waves along N
  const int lane = tid & 31;

  v8f acc[2][4];
#pragma unroll
  for (int i = 0; i < 2; ++i)
#pragma unroll
    for (int j = 0; j < 4; ++j)
#pragma unroll
      for (int e = 0; e < 8; ++e) acc[i][j][e] = 0.f;

  const int lr = tid >> 1;           // 0..127
  const int lc = (tid & 1) * 16;     // 0 or 16
  const _Float16* Ag = A + (size_t)(m0 + lr) * K + lc;
  const _Float16* Wg = W + (size_t)(n0 + lr) * K + lc;
  uint32_t ldsA[2], ldsB[2];
#pragma unroll
  for (int b = 0; b < 2; ++b) {
    ldsA[b] = (uint32_t)(uintptr_t)&As[b][lr * LDA + lc];
    ldsB[b] = (uint32_t)(uintptr_t)&Bs[b][lr * LDA + lc];
  }

  auto issue = [&](int buf, int k0) {
    async_copy32(ldsA[buf], Ag + k0);
    async_copy32(ldsB[buf], Wg + k0);
  };

  const int nk = K / BK;
  issue(0, 0);
  for (int kt = 0; kt < nk; ++kt) {
    const int cur = kt & 1;
    wait_async0();        // own writes into `cur` complete
    __syncthreads();      // all waves: `cur` written; `cur^1` fully read
    if (kt + 1 < nk) issue(cur ^ 1, (kt + 1) * BK);

    v16h af[2], bf[4];
    af[0] = load_frag(&As[cur][(wm + 0)  * LDA], LDA);
    af[1] = load_frag(&As[cur][(wm + 16) * LDA], LDA);
#pragma unroll
    for (int j = 0; j < 4; ++j)
      bf[j] = load_frag(&Bs[cur][(wn + j * 16) * LDA], LDA);
#pragma unroll
    for (int i = 0; i < 2; ++i)
#pragma unroll
      for (int j = 0; j < 4; ++j) acc[i][j] = wmma32(af[i], bf[j], acc[i][j]);
  }

  const int crow = lane >> 4;
  const int ccol = lane & 15;
#pragma unroll
  for (int i = 0; i < 2; ++i)
#pragma unroll
    for (int j = 0; j < 4; ++j) {
      int col = n0 + wn + j * 16 + ccol;
      float bj = bias[col];
#pragma unroll
      for (int v = 0; v < 8; ++v) {
        int row = m0 + wm + i * 16 + v + 8 * crow;
        size_t idx = (size_t)row * N + col;
        float val = acc[i][j][v] + bj;
        if (MODE == 0) {
          outH[idx] = (_Float16)val;
        } else if (MODE == 1) {
          outF[idx] = resid[idx] + val;
        } else {
          outH[idx] = (_Float16)(0.5f * val * (1.f + erff(val * 0.70710678f)));
        }
      }
    }
}

// ---------------------------------------------------------------- attention
// grid (qblock=8, head=16, batch=8), 256 thr = 8 waves; wave owns 16 queries.
// K chunk (128x64) and V^T chunk (64x128) in LDS; online softmax.

__global__ __launch_bounds__(256)
void attn_kernel(const _Float16* __restrict__ Q, const _Float16* __restrict__ Kh,
                 const _Float16* __restrict__ Vh, _Float16* __restrict__ O) {
  __shared__ __align__(16) _Float16 Ks[128 * 72];
  __shared__ __align__(16) _Float16 Vt[64 * 136];
  __shared__ __align__(16) _Float16 Pst[8][16 * 32];

  const int qb = blockIdx.x, hh = blockIdx.y, bb = blockIdx.z;
  const int tid = threadIdx.x;
  const int w = tid >> 5;
  const int lane = tid & 31;
  const int q0 = qb * 128 + w * 16;
  const size_t base = ((size_t)bb * TT) * EE + (size_t)hh * DD;

  v16h aQ0 = load_frag(Q + base + (size_t)q0 * EE, EE);       // d = 0..31
  v16h aQ1 = load_frag(Q + base + (size_t)q0 * EE + 32, EE);  // d = 32..63

  float mrow[8], lrow[8];
  v8f accO[4];
#pragma unroll
  for (int v = 0; v < 8; ++v) { mrow[v] = -1e30f; lrow[v] = 0.f; }
#pragma unroll
  for (int j = 0; j < 4; ++j)
#pragma unroll
    for (int v = 0; v < 8; ++v) accO[j][v] = 0.f;

  for (int sc = 0; sc < 8; ++sc) {
    __syncthreads();   // protect Ks/Vt WAR vs previous chunk
    {
      int r = tid >> 1, part = tid & 1;
      const _Float16* ks = Kh + base + (size_t)(sc * 128 + r) * EE + part * 32;
#pragma unroll
      for (int c = 0; c < 4; ++c)
        *(half8*)&Ks[r * 72 + part * 32 + c * 8] = *(const half8*)(ks + c * 8);
      const _Float16* vs = Vh + base + (size_t)(sc * 128 + r) * EE + part * 32;
#pragma unroll
      for (int c = 0; c < 32; ++c)
        Vt[(part * 32 + c) * 136 + r] = vs[c];   // transpose: Vt[d][s]
    }
    __syncthreads();

    // S = (Q K^T) * 1/sqrt(D), 16 queries x 128 keys in 8 fragments
    v8f sf[8];
#pragma unroll
    for (int n = 0; n < 8; ++n) {
      v16h b0 = load_frag(&Ks[(n * 16) * 72], 72);
      v16h b1 = load_frag(&Ks[(n * 16) * 72 + 32], 72);
      v8f c;
#pragma unroll
      for (int e = 0; e < 8; ++e) c[e] = 0.f;
      c = wmma32(aQ0, b0, c);
      c = wmma32(aQ1, b1, c);
#pragma unroll
      for (int v = 0; v < 8; ++v) sf[n][v] = c[v] * 0.125f;
    }

    // online softmax: rows live in (VGPR v, lane-half); reduce across 16 lanes
#pragma unroll
    for (int v = 0; v < 8; ++v) {
      float mx = sf[0][v];
#pragma unroll
      for (int n = 1; n < 8; ++n) mx = fmaxf(mx, sf[n][v]);
#pragma unroll
      for (int m = 1; m <= 8; m <<= 1) mx = fmaxf(mx, __shfl_xor(mx, m, 32));
      float mnew = fmaxf(mrow[v], mx);
      float corr = __expf(mrow[v] - mnew);
      mrow[v] = mnew;
      float rsum = 0.f;
#pragma unroll
      for (int n = 0; n < 8; ++n) {
        float p = __expf(sf[n][v] - mnew);
        sf[n][v] = p;
        rsum += p;
      }
#pragma unroll
      for (int m = 1; m <= 8; m <<= 1) rsum += __shfl_xor(rsum, m, 32);
      lrow[v] = lrow[v] * corr + rsum;
#pragma unroll
      for (int j = 0; j < 4; ++j) accO[j][v] *= corr;
    }

    // O += P V  (re-lay P into A fragments through per-wave LDS staging)
#pragma unroll
    for (int p = 0; p < 4; ++p) {
#pragma unroll
      for (int f = 0; f < 2; ++f)
#pragma unroll
        for (int v = 0; v < 8; ++v)
          Pst[w][(v + 8 * (lane >> 4)) * 32 + f * 16 + (lane & 15)] =
              (_Float16)sf[p * 2 + f][v];
      __syncthreads();
      v16h aP = load_frag(&Pst[w][0], 32);
#pragma unroll
      for (int j = 0; j < 4; ++j) {
        v16h bV = load_frag(&Vt[(j * 16) * 136 + p * 32], 136);
        accO[j] = wmma32(aP, bV, accO[j]);
      }
      __syncthreads();
    }
  }

#pragma unroll
  for (int j = 0; j < 4; ++j)
#pragma unroll
    for (int v = 0; v < 8; ++v) {
      int qq = q0 + v + 8 * (lane >> 4);
      int d = j * 16 + (lane & 15);
      O[base + (size_t)qq * EE + d] = (_Float16)(accO[j][v] / lrow[v]);
    }
}

// ---------------------------------------------------------------- head/decode

__global__ void head_kernel(const _Float16* __restrict__ h,
                            const float* __restrict__ hw,
                            float* __restrict__ logits) {
  int t = blockIdx.x * 8 + (threadIdx.x >> 5);
  int c = threadIdx.x & 31;
  const _Float16* hr = h + (size_t)t * EE;
  const float* wr = hw + (size_t)c * EE;
  float acc = 0.f;
  for (int e = 0; e < EE; ++e) acc += (float)hr[e] * wr[e];
  logits[(size_t)t * 32 + c] = acc;
}

__global__ void decode_kernel(const float* __restrict__ logits,
                              const float* __restrict__ pcomp,
                              const float* __restrict__ pmean,
                              float* __restrict__ out) {
  int id = blockIdx.x * 256 + threadIdx.x;   // over 8*256*256
  int b = id >> 16, rem = id & 65535;
  int yy = rem >> 8, xx = rem & 255;
  int hh = yy >> 3, p1 = yy & 7, ww = xx >> 3, p2 = xx & 7;
  int t = (b * 32 + hh) * 32 + ww;
  int j = p1 * 8 + p2;
  float acc = pmean[j];
#pragma unroll
  for (int c = 0; c < 32; ++c)
    acc += logits[(size_t)t * 32 + c] * pcomp[c * 64 + j];
  out[id] = acc;
}

// ---------------------------------------------------------------- launch

extern "C" void kernel_launch(void* const* d_in, const int* in_sizes, int n_in,
                              void* d_out, int out_size, void* d_ws, size_t ws_size,
                              hipStream_t stream) {
  const float* img   = (const float*)d_in[0];
  const float* mmean = (const float*)d_in[1];
  const float* mcomp = (const float*)d_in[2];
  const float* pcomp = (const float*)d_in[3];
  const float* pmean = (const float*)d_in[4];
  const float* tokw  = (const float*)d_in[5];
  const float* pos   = (const float*)d_in[6];
  const float* ln1w  = (const float*)d_in[7];
  const float* ln1b  = (const float*)d_in[8];
  const float* wq    = (const float*)d_in[9];
  const float* bq    = (const float*)d_in[10];
  const float* wk    = (const float*)d_in[11];
  const float* bk    = (const float*)d_in[12];
  const float* wv    = (const float*)d_in[13];
  const float* bv    = (const float*)d_in[14];
  const float* wp    = (const float*)d_in[15];
  const float* bp    = (const float*)d_in[16];
  const float* ln2w  = (const float*)d_in[17];
  const float* ln2b  = (const float*)d_in[18];
  const float* w1    = (const float*)d_in[19];
  const float* b1    = (const float*)d_in[20];
  const float* w2    = (const float*)d_in[21];
  const float* b2    = (const float*)d_in[22];
  const float* lnfw  = (const float*)d_in[23];
  const float* lnfb  = (const float*)d_in[24];
  const float* headw = (const float*)d_in[25];

  char* ws = (char*)d_ws;
  size_t off = 0;
  auto carve = [&](size_t bytes) {
    void* p = ws + off;
    off += (bytes + 255) & ~(size_t)255;
    return p;
  };
  float*    x      = (float*)   carve((size_t)MTOT * EE * 4);    // residual stream
  _Float16* h      = (_Float16*)carve((size_t)MTOT * EE * 2);    // LN output
  _Float16* qh     = (_Float16*)carve((size_t)MTOT * EE * 2);
  _Float16* kh     = (_Float16*)carve((size_t)MTOT * EE * 2);
  _Float16* vh     = (_Float16*)carve((size_t)MTOT * EE * 2);
  _Float16* yh     = (_Float16*)carve((size_t)MTOT * EE * 2);    // attn out
  _Float16* mh     = (_Float16*)carve((size_t)MTOT * FF_ * 2);   // MLP hidden
  _Float16* wbuf   = (_Float16*)carve((size_t)FF_ * EE * 2);     // f16 weights
  float*    coef   = (float*)   carve((size_t)MTOT * 32 * 4);
  float*    logits = (float*)   carve((size_t)MTOT * 32 * 4);
  (void)ws_size; (void)in_sizes; (void)n_in; (void)out_size;

  const int nEE = EE * EE;        // 1 Mi elements
  const int nFE = FF_ * EE;       // 4 Mi elements

  patch_pca_kernel<<<(MTOT * 32) / 256, 256, 0, stream>>>(img, mmean, mcomp, coef);
  embed_kernel<<<(MTOT * EE) / 256, 256, 0, stream>>>(coef, tokw, pos, x);

  for (int l = 0; l < 12; ++l) {
    ln_kernel<<<MTOT, 256, 0, stream>>>(x, ln1w + l * EE, ln1b + l * EE, h);

    f32tof16_kernel<<<nEE / 256, 256, 0, stream>>>(wq + (size_t)l * nEE, wbuf, nEE);
    gemm_kernel<0><<<dim3(MTOT / BM, EE / BN), 256, 0, stream>>>(
        h, wbuf, bq + l * EE, nullptr, nullptr, qh, EE, EE);
    f32tof16_kernel<<<nEE / 256, 256, 0, stream>>>(wk + (size_t)l * nEE, wbuf, nEE);
    gemm_kernel<0><<<dim3(MTOT / BM, EE / BN), 256, 0, stream>>>(
        h, wbuf, bk + l * EE, nullptr, nullptr, kh, EE, EE);
    f32tof16_kernel<<<nEE / 256, 256, 0, stream>>>(wv + (size_t)l * nEE, wbuf, nEE);
    gemm_kernel<0><<<dim3(MTOT / BM, EE / BN), 256, 0, stream>>>(
        h, wbuf, bv + l * EE, nullptr, nullptr, vh, EE, EE);

    attn_kernel<<<dim3(8, HH, BB), 256, 0, stream>>>(qh, kh, vh, yh);

    f32tof16_kernel<<<nEE / 256, 256, 0, stream>>>(wp + (size_t)l * nEE, wbuf, nEE);
    gemm_kernel<1><<<dim3(MTOT / BM, EE / BN), 256, 0, stream>>>(
        yh, wbuf, bp + l * EE, x, x, nullptr, EE, EE);

    ln_kernel<<<MTOT, 256, 0, stream>>>(x, ln2w + l * EE, ln2b + l * EE, h);

    f32tof16_kernel<<<nFE / 256, 256, 0, stream>>>(w1 + (size_t)l * nFE, wbuf, nFE);
    gemm_kernel<2><<<dim3(MTOT / BM, FF_ / BN), 256, 0, stream>>>(
        h, wbuf, b1 + l * FF_, nullptr, nullptr, mh, EE, FF_);

    f32tof16_kernel<<<nFE / 256, 256, 0, stream>>>(w2 + (size_t)l * nFE, wbuf, nFE);
    gemm_kernel<1><<<dim3(MTOT / BM, EE / BN), 256, 0, stream>>>(
        mh, wbuf, b2 + l * EE, x, x, nullptr, FF_, EE);
  }

  ln_kernel<<<MTOT, 256, 0, stream>>>(x, lnfw, lnfb, h);
  head_kernel<<<MTOT / 8, 256, 0, stream>>>(h, headw, logits);
  decode_kernel<<<(BB * 256 * 256) / 256, 256, 0, stream>>>(
      logits, pcomp, pmean, (float*)d_out);
}
